// GatingMixedDecoder_2095944040863
// MI455X (gfx1250) — compile-verified, matrix operand order checked by reading, not run on hardware
//
#include <hip/hip_runtime.h>
#include <math.h>

typedef __bf16 bf16_t;
typedef __attribute__((ext_vector_type(16))) __bf16 v16bf;
typedef __attribute__((ext_vector_type(8)))  __bf16 v8bf;
typedef __attribute__((ext_vector_type(8)))  float  v8f;

constexpr int NB   = 16384;  // batch
constexpr int SDIM = 512;
constexpr int ADIM = 512;
constexpr int HDIM = 1024;
constexpr int LDIM = 64;
constexpr int NE   = 6;

// ---------------------------------------------------------------- utilities

__global__ void cvt_bf16(const float* __restrict__ src, bf16_t* __restrict__ dst, int n) {
    for (int i = blockIdx.x * blockDim.x + threadIdx.x; i < n; i += gridDim.x * blockDim.x)
        dst[i] = (bf16_t)src[i];
}

// Convert weights f32 [E, J, K] -> bf16 in WMMA-B-fragment swizzled layout:
// block index = (e*(J/32) + j/32)*(K/16) + n/16 ; within block: lane*16 + i,
// lane = (n&15) + 16*((j&8)>>3), i = (j&7) | ((j&16)>>1)   (ISA 16-bit 32x16 layout)
__global__ void swz_w(const float* __restrict__ w, bf16_t* __restrict__ dst,
                      int J, int K, int total) {
    for (int idx = blockIdx.x * blockDim.x + threadIdx.x; idx < total;
         idx += gridDim.x * blockDim.x) {
        int n = idx % K;
        int t = idx / K;
        int j = t % J;
        int e = t / J;
        int kc   = j >> 5;
        int jr   = j & 31;
        int nt   = n >> 4;
        int lane = (n & 15) + (((jr & 8) >> 3) << 4);
        int ii   = (jr & 7) | ((jr & 16) >> 1);
        size_t blk = (size_t)(e * (J >> 5) + kc) * (size_t)(K >> 4) + nt;
        dst[blk * 512 + lane * 16 + ii] = (bf16_t)w[idx];
    }
}

// deterministic hash-based standard normal (stand-in for jax fixed-key noise;
// compile-only environment, no bit-exactness check possible)
__device__ __forceinline__ float gnoise(unsigned idx) {
    unsigned s1 = idx * 747796405u + 2891336453u;
    unsigned w1 = ((s1 >> ((s1 >> 28) + 4u)) ^ s1) * 277803737u; w1 = (w1 >> 22) ^ w1;
    unsigned s2 = (idx ^ 0x9E3779B9u) * 747796405u + 2891336453u;
    unsigned w2 = ((s2 >> ((s2 >> 28) + 4u)) ^ s2) * 277803737u; w2 = (w2 >> 22) ^ w2;
    float u1 = ((w1 >> 9) + 1u) * (1.0f / 8388608.0f);   // (0,1]
    float u2 = (w2 >> 8) * (1.0f / 16777216.0f);         // [0,1)
    return sqrtf(-2.0f * __logf(u1)) * __cosf(6.28318530718f * u2);
}

// ---------------------------------------------------------------- gating MLP
// one wave32 per batch row; lane = hidden unit (G == 32)
__global__ __launch_bounds__(256) void gating(
        const float* __restrict__ state, const float* __restrict__ latent,
        const float* __restrict__ gw0, const float* __restrict__ gb0,
        const float* __restrict__ gw1, const float* __restrict__ gb1,
        const float* __restrict__ gw2, const float* __restrict__ gb2,
        float* __restrict__ coef) {
    const int lane = threadIdx.x & 31;
    const int wav  = threadIdx.x >> 5;
    const int b    = blockIdx.x * 8 + wav;
    const float* sb = state + (size_t)b * SDIM;
    const float* lb = latent + (size_t)b * LDIM;

    float h1 = gb0[lane];
#pragma unroll 4
    for (int j = 0; j < SDIM; ++j) h1 = fmaf(sb[j], gw0[j * 32 + lane], h1);
#pragma unroll 4
    for (int j = 0; j < LDIM; ++j) h1 = fmaf(lb[j], gw0[(SDIM + j) * 32 + lane], h1);
    h1 = h1 > 0.f ? h1 : expm1f(h1);

    float h2 = gb1[lane];
#pragma unroll
    for (int m = 0; m < 32; ++m) h2 = fmaf(__shfl(h1, m, 32), gw1[m * 32 + lane], h2);
    h2 = h2 > 0.f ? h2 : expm1f(h2);

    float lg[NE];
#pragma unroll
    for (int e = 0; e < NE; ++e) lg[e] = gb2[e];
    for (int m = 0; m < 32; ++m) {
        float hm = __shfl(h2, m, 32);
#pragma unroll
        for (int e = 0; e < NE; ++e) lg[e] = fmaf(hm, gw2[m * NE + e], lg[e]);
    }
    float mx = lg[0];
#pragma unroll
    for (int e = 1; e < NE; ++e) mx = fmaxf(mx, lg[e]);
    float s = 0.f;
#pragma unroll
    for (int e = 0; e < NE; ++e) { lg[e] = __expf(lg[e] - mx); s += lg[e]; }
    float inv = 1.0f / s;
    if (lane < NE) coef[(size_t)b * NE + lane] = lg[lane] * inv;
}

// ---------------------------------------------------------------- mixture GEMM
// out[b,k] = sum_e coef[b,e] * ( concat(act,latent)[b,:] @ W_e )[k] + (coef @ bias)[b,k]
// Block tile 128x128, 512 threads = 16 waves as 4(M) x 4(N), wave tile 32x32 =
// 2x2 WMMA frags, bf16 WMMA 16x16x32 with f32 accumulate, LDS double buffering.
// Expert-outer loop with temp accumulator, mixed into final acc by per-row coef.
__global__ __launch_bounds__(512) void mix_gemm(
        const bf16_t* __restrict__ act,    // [B, J1] bf16
        const bf16_t* __restrict__ latbf,  // [B, 64] bf16
        const bf16_t* __restrict__ wswz,   // swizzled bf16 weights
        const float*  __restrict__ bias,   // [E, K]
        const float*  __restrict__ coef,   // [B, E]
        bf16_t* __restrict__ out_bf,       // next-layer activation (mode 0)
        float*  __restrict__ out_f32,      // final output (mode 1)
        int J1, int K, int mode) {
    constexpr int BM = 128, BN = 128, ASTRIDE = 40, NT = 8;  // 8 n-subtiles of 16
    const int NK  = (J1 + LDIM) >> 5;   // K-chunks of 32
    const int KC  = K >> 4;             // 16-col subtiles per (e,kc) in swizzled weights
    const int n0  = blockIdx.x * BN;
    const int nt0 = blockIdx.x * NT;
    const int m0  = blockIdx.y * BM;

    __shared__ __align__(32) bf16_t Asub[2][BM * ASTRIDE];
    __shared__ __align__(32) bf16_t Bsub[2][NT * 512];
    __shared__ float Cf[BM * 8];

    const int tid  = threadIdx.x;
    const int lane = tid & 31;
    const int wav  = tid >> 5;      // 0..15
    const int wm   = wav >> 2;      // 0..3  (M)
    const int wn   = wav & 3;       // 0..3  (N)
    const int l15  = lane & 15;
    const int lhi  = lane >> 4;     // K-half select for A frags

    // stage coef tile into LDS (padded stride 8)
    for (int i = tid; i < BM * NE; i += 512) {
        int r = i / NE, e = i - r * NE;
        Cf[r * 8 + e] = coef[(size_t)(m0 + r) * NE + e];
    }

    v8f facc[2][2];
#pragma unroll
    for (int a = 0; a < 2; ++a)
#pragma unroll
        for (int c = 0; c < 2; ++c)
            facc[a][c] = (v8f){0.f, 0.f, 0.f, 0.f, 0.f, 0.f, 0.f, 0.f};

    // per-thread staging coordinates: exactly one v8bf per thread for A and B
    const int arow = tid >> 2;      // 0..127
    const int acv  = tid & 3;       // 8-element column chunk within 32
    const int bst  = tid >> 6;      // 0..7 n-subtile
    const int bv   = tid & 63;      // 8-element chunk within 512

    for (int e = 0; e < NE; ++e) {
        v8f tacc[2][2];
#pragma unroll
        for (int a = 0; a < 2; ++a)
#pragma unroll
            for (int c = 0; c < 2; ++c)
                tacc[a][c] = (v8f){0.f, 0.f, 0.f, 0.f, 0.f, 0.f, 0.f, 0.f};

        const bf16_t* wbase = wswz + (size_t)e * (size_t)((J1 + LDIM) >> 5) * KC * 512;

        auto aptr = [&](int j0) -> const bf16_t* {
            return (j0 < J1) ? act   + (size_t)(m0 + arow) * J1   + j0 + acv * 8
                             : latbf + (size_t)(m0 + arow) * LDIM + (j0 - J1) + acv * 8;
        };

        // stage kc = 0 (always inside the act region: J1 >= 32)
        v8bf ra = *(const v8bf*)aptr(0);
        v8bf rb = *(const v8bf*)(wbase + (size_t)(nt0 + bst) * 512 + bv * 8);
        *(v8bf*)(&Asub[0][arow * ASTRIDE + acv * 8]) = ra;
        *(v8bf*)(&Bsub[0][bst * 512 + bv * 8])       = rb;
        __syncthreads();

        for (int kc = 0; kc < NK; ++kc) {
            const int  cur  = kc & 1;
            const bool more = (kc + 1) < NK;
            if (more) {
                const int j0 = (kc + 1) << 5;
                ra = *(const v8bf*)aptr(j0);
                rb = *(const v8bf*)(wbase + ((size_t)(kc + 1) * KC + nt0 + bst) * 512 + bv * 8);
                if (kc + 2 < NK)
                    __builtin_prefetch(wbase + ((size_t)(kc + 2) * KC + nt0 + bst) * 512 + bv * 8, 0, 1);
            }

            // A fragments: lane<16 -> K {0..7,16..23}; lane>=16 -> K {8..15,24..31}
            v16bf afr[2], bfr[2];
#pragma unroll
            for (int fm = 0; fm < 2; ++fm) {
                const bf16_t* pa = &Asub[cur][(wm * 32 + fm * 16 + l15) * ASTRIDE + lhi * 8];
                v8bf lo = *(const v8bf*)pa;
                v8bf hi = *(const v8bf*)(pa + 16);
                afr[fm] = __builtin_shufflevector(lo, hi, 0, 1, 2, 3, 4, 5, 6, 7,
                                                  8, 9, 10, 11, 12, 13, 14, 15);
            }
#pragma unroll
            for (int fn = 0; fn < 2; ++fn)
                bfr[fn] = *(const v16bf*)(&Bsub[cur][(wn * 2 + fn) * 512 + lane * 16]);

#pragma unroll
            for (int fm = 0; fm < 2; ++fm)
#pragma unroll
                for (int fn = 0; fn < 2; ++fn)
                    tacc[fm][fn] = __builtin_amdgcn_wmma_f32_16x16x32_bf16(
                        false, afr[fm], false, bfr[fn], (short)0, tacc[fm][fn],
                        false, false);

            if (more) {
                const int nxt = cur ^ 1;
                *(v8bf*)(&Asub[nxt][arow * ASTRIDE + acv * 8]) = ra;
                *(v8bf*)(&Bsub[nxt][bst * 512 + bv * 8])       = rb;
            }
            __syncthreads();
        }

        // mix this expert into the final accumulator: C layout row = base + r + 8*lhi
#pragma unroll
        for (int fm = 0; fm < 2; ++fm)
#pragma unroll
            for (int r = 0; r < 8; ++r) {
                float c = Cf[(wm * 32 + fm * 16 + lhi * 8 + r) * 8 + e];
                facc[fm][0][r] = fmaf(c, tacc[fm][0][r], facc[fm][0][r]);
                facc[fm][1][r] = fmaf(c, tacc[fm][1][r], facc[fm][1][r]);
            }
    }

    // epilogue: + coef @ bias, activation / noise, store
#pragma unroll
    for (int fn = 0; fn < 2; ++fn) {
        const int col = n0 + wn * 32 + fn * 16 + l15;
        float bv6[NE];
#pragma unroll
        for (int e = 0; e < NE; ++e) bv6[e] = bias[e * K + col];
#pragma unroll
        for (int fm = 0; fm < 2; ++fm)
#pragma unroll
            for (int r = 0; r < 8; ++r) {
                const int rl = wm * 32 + fm * 16 + lhi * 8 + r;
                float mb = 0.f;
#pragma unroll
                for (int e = 0; e < NE; ++e) mb = fmaf(Cf[rl * 8 + e], bv6[e], mb);
                float v = facc[fm][fn][r] + mb;
                const size_t oi = (size_t)(m0 + rl) * K + col;
                if (mode == 0) {
                    v = v > 0.f ? v : expm1f(v);
                    out_bf[oi] = (bf16_t)v;
                } else {
                    out_f32[oi] = v + 0.05f * gnoise((unsigned)oi);
                }
            }
    }
}

// ---------------------------------------------------------------- launch

extern "C" void kernel_launch(void* const* d_in, const int* in_sizes, int n_in,
                              void* d_out, int out_size, void* d_ws, size_t ws_size,
                              hipStream_t stream) {
    (void)in_sizes; (void)n_in; (void)out_size; (void)ws_size;
    const float* state  = (const float*)d_in[0];
    const float* latent = (const float*)d_in[1];
    const float* w[4]   = {(const float*)d_in[2], (const float*)d_in[4],
                           (const float*)d_in[6], (const float*)d_in[8]};
    const float* bs[4]  = {(const float*)d_in[3], (const float*)d_in[5],
                           (const float*)d_in[7], (const float*)d_in[9]};
    const float* gw0 = (const float*)d_in[10]; const float* gb0 = (const float*)d_in[11];
    const float* gw1 = (const float*)d_in[12]; const float* gb1 = (const float*)d_in[13];
    const float* gw2 = (const float*)d_in[14]; const float* gb2 = (const float*)d_in[15];

    // workspace partition (~127 MB total)
    char* base = (char*)d_ws; size_t off = 0;
    auto take = [&](size_t bytes) -> void* {
        void* r = base + off;
        off = (off + bytes + 255) & ~(size_t)255;
        return r;
    };
    float*  coef     = (float*) take((size_t)NB * NE * 4);
    bf16_t* state_bf = (bf16_t*)take((size_t)NB * SDIM * 2);
    bf16_t* lat_bf   = (bf16_t*)take((size_t)NB * LDIM * 2);
    bf16_t* actA     = (bf16_t*)take((size_t)NB * HDIM * 2);
    bf16_t* actB     = (bf16_t*)take((size_t)NB * HDIM * 2);
    bf16_t* wsw0     = (bf16_t*)take((size_t)NE * 576  * 1024 * 2);
    bf16_t* wsw1     = (bf16_t*)take((size_t)NE * 1088 * 1024 * 2);
    bf16_t* wsw2     = (bf16_t*)take((size_t)NE * 1088 * 1024 * 2);
    bf16_t* wsw3     = (bf16_t*)take((size_t)NE * 1088 * 512  * 2);

    cvt_bf16<<<2048, 256, 0, stream>>>(state,  state_bf, NB * SDIM);
    cvt_bf16<<<512,  256, 0, stream>>>(latent, lat_bf,   NB * LDIM);

    swz_w<<<4096, 256, 0, stream>>>(w[0], wsw0, 576,  1024, NE * 576  * 1024);
    swz_w<<<4096, 256, 0, stream>>>(w[1], wsw1, 1088, 1024, NE * 1088 * 1024);
    swz_w<<<4096, 256, 0, stream>>>(w[2], wsw2, 1088, 1024, NE * 1088 * 1024);
    swz_w<<<4096, 256, 0, stream>>>(w[3], wsw3, 1088, 512,  NE * 1088 * 512);

    gating<<<NB / 8, 256, 0, stream>>>(state, latent, gw0, gb0, gw1, gb1, gw2, gb2, coef);

    dim3 blk(512);
    mix_gemm<<<dim3(HDIM / 128, NB / 128), blk, 0, stream>>>(
        state_bf, lat_bf, wsw0, bs[0], coef, actA, nullptr, SDIM, HDIM, 0);
    mix_gemm<<<dim3(HDIM / 128, NB / 128), blk, 0, stream>>>(
        actA, lat_bf, wsw1, bs[1], coef, actB, nullptr, HDIM, HDIM, 0);
    mix_gemm<<<dim3(HDIM / 128, NB / 128), blk, 0, stream>>>(
        actB, lat_bf, wsw2, bs[2], coef, actA, nullptr, HDIM, HDIM, 0);
    mix_gemm<<<dim3(ADIM / 128, NB / 128), blk, 0, stream>>>(
        actA, lat_bf, wsw3, bs[3], coef, nullptr, (float*)d_out, HDIM, ADIM, 1);
}
